// DSModelMultiQ_6055903887802
// MI455X (gfx1250) — matrix operand order repaired and checked by reading, not compiled, not run on hardware
//
#include <hip/hip_runtime.h>

// ---------------------------------------------------------------------------
// DS-model belief computation for MI455X (gfx1250, wave32, WMMA).
//
// B=8192 samples, F=128 feats, L=4096 literals, R=8192 rules, K=64.
// lit2rule has exactly 3 nonzeros per rule column -> fired[b,r] is an AND of
// 3 literal truth bits (no 550-GFLOP dense matmul). The remaining
// fired @ [log_mw | log_w] GEMM (contraction over R=8192) runs on
// v_wmma_f32_16x16x32_f16; fired is kept as a bitmask end-to-end and expanded
// to {0,1} f16 A-fragments with pure bit-selects in the GEMM inner loop.
// ---------------------------------------------------------------------------

typedef __attribute__((ext_vector_type(16))) _Float16 v16h;
typedef __attribute__((ext_vector_type(8)))  float    v8f;

#define B_N   8192
#define F_N   128
#define L_N   4096
#define R_N   8192
#define K_N   64
#define NPAD  80          // K cols (0..63) + log_w col (64) + zero pad (65..79)
#define SPAD  84          // LDS row stride (floats), bank-conflict padding
#define EPSF  1e-12f

// ---------------- kernel 1: zero rule counters ----------------
__global__ void zero_counts(unsigned* __restrict__ cnt) {
    int r = blockIdx.x * 256 + threadIdx.x;
    if (r < R_N) cnt[r] = 0u;
}

// ---------------- kernel 2: literal truth bitmask (B x L/32 words) ---------
__global__ void truth_kernel(const float* __restrict__ X,
                             const float* __restrict__ lit_value,
                             const int*   __restrict__ feat,
                             const int*   __restrict__ opc,
                             unsigned*    __restrict__ truthBits) {
    int gid = blockIdx.x * 256 + threadIdx.x;       // B * (L/32) threads
    int b   = gid >> 7;                             // L/32 == 128 words/row
    int wrd = gid & 127;
    const float* xrow = X + b * F_N;
    unsigned bits = 0u;
    int l0 = wrd * 32;
    #pragma unroll 4
    for (int i = 0; i < 32; ++i) {
        int   l = l0 + i;
        float x = xrow[feat[l]];
        float v = lit_value[l];
        int   o = opc[l];
        bool  t = (o == 0) ? (x == v) : ((o == 1) ? (x < v) : (x > v));
        bits |= (t ? 1u : 0u) << i;
    }
    truthBits[gid] = bits;
}

// ---------------- kernel 3: recover per-rule literal ids from lit2rule -----
__global__ void extract_kernel(const float* __restrict__ lit2rule,
                               unsigned* __restrict__ cnt,
                               unsigned* __restrict__ tmp) {
    unsigned idx = blockIdx.x * 256u + threadIdx.x;  // L*R entries, coalesced in r
    int r = (int)(idx & (R_N - 1));
    int l = (int)(idx >> 13);                        // / R_N
    if (lit2rule[idx] != 0.0f) {
        unsigned s = atomicAdd(&cnt[r], 1u);
        if (s < 3u) tmp[r * 3 + s] = (unsigned)l;
    }
}

// ---------------- kernel 4: pack 3 literal ids into uint2 per rule ---------
__global__ void pack_kernel(const unsigned* __restrict__ cnt,
                            const unsigned* __restrict__ tmp,
                            uint2* __restrict__ packed) {
    int r = blockIdx.x * 256 + threadIdx.x;
    if (r >= R_N) return;
    unsigned c = cnt[r]; if (c > 3u) c = 3u;
    unsigned a = (c > 0u) ? tmp[r * 3 + 0] : 0u;
    unsigned b = (c > 1u) ? tmp[r * 3 + 1] : a;   // pad with duplicates:
    unsigned d = (c > 2u) ? tmp[r * 3 + 2] : a;   // AND is idempotent
    packed[r] = make_uint2(a | (b << 16), d);
}

// ---------------- kernel 5: softmax -> transposed padded f16 log-mass ------
// BmatT layout: [n][r], n in 0..79.  n<64: log(m_n+omega+eps); n==64: log(omega+eps);
// n>64: 0.  Transposed so B-fragment lanes load 16 contiguous halves.
__global__ void softmax_kernel(const float* __restrict__ params,
                               _Float16* __restrict__ BmatT) {
    int r = blockIdx.x * 256 + threadIdx.x;
    if (r >= R_N) return;
    const float* p = params + r * (K_N + 1);
    float mx = p[0];
    for (int k = 1; k <= K_N; ++k) mx = fmaxf(mx, p[k]);
    float denom = 0.0f;
    for (int k = 0; k <= K_N; ++k) denom += expf(p[k] - mx);
    float inv   = 1.0f / denom;
    float omega = expf(p[K_N] - mx) * inv;
    for (int n = 0; n < K_N; ++n) {
        float m = expf(p[n] - mx) * inv;
        BmatT[n * R_N + r] = (_Float16)logf(m + omega + EPSF);
    }
    BmatT[K_N * R_N + r] = (_Float16)logf(omega + EPSF);
    for (int n = K_N + 1; n < NPAD; ++n) BmatT[n * R_N + r] = (_Float16)0.0f;
}

// ---------------- kernel 6: fired bitmask (B x R/32 words) -----------------
// One block per sample: truth row (512 B) cached in LDS; each thread ANDs
// 3 bit-tests for each of its 32 rules into one output word.
__global__ __launch_bounds__(256)
void fired_kernel(const unsigned* __restrict__ truthBits,
                  const uint2*    __restrict__ packedLits,
                  unsigned*       __restrict__ firedWords) {
    __shared__ unsigned trow[128];
    int b = blockIdx.x;
    if (threadIdx.x < 128) trow[threadIdx.x] = truthBits[b * 128 + threadIdx.x];
    __syncthreads();
    int w = threadIdx.x;                 // 256 words of 32 rules each
    int rbase = w * 32;
    unsigned bits = 0u;
    #pragma unroll 4
    for (int i = 0; i < 32; ++i) {
        uint2 pl = packedLits[rbase + i];
        unsigned l0 = pl.x & 0xFFFFu, l1 = pl.x >> 16, l2 = pl.y;
        unsigned t = (trow[l0 >> 5] >> (l0 & 31u)) &
                     (trow[l1 >> 5] >> (l1 & 31u)) &
                     (trow[l2 >> 5] >> (l2 & 31u)) & 1u;
        bits |= t << i;
    }
    firedWords[b * 256 + w] = bits;
}

// ---------------- kernel 7: WMMA GEMM + epilogue ---------------------------
// One wave owns 32 samples (2 M-tiles) x all 80 output cols:
// 2x5 f32 accumulators, each B fragment feeds 2 WMMAs.
__global__ __launch_bounds__(128)
void ds_gemm_kernel(const unsigned* __restrict__ firedWords,
                    const _Float16* __restrict__ BmatT,
                    const float*    __restrict__ prior,
                    float*          __restrict__ out) {
    __shared__ float ldsS[4][32][SPAD];          // 43 KB result staging

    const int wave = threadIdx.x >> 5;
    const int lane = threadIdx.x & 31;
    const int m    = lane & 15;                  // row within a 16-sample tile
    const int hi   = lane >> 4;                  // lane half (A/B K-split)
    const int sampleBase = (blockIdx.x * 4 + wave) * 32;

    const v8f z = {0.f, 0.f, 0.f, 0.f, 0.f, 0.f, 0.f, 0.f};
    v8f acc[2][5] = {{z, z, z, z, z}, {z, z, z, z, z}};

    const unsigned* fw0 = firedWords + (size_t)(sampleBase + m)      * 256;
    const unsigned* fw1 = firedWords + (size_t)(sampleBase + 16 + m) * 256;

    typedef union { v16h h; unsigned u[8]; } AF;

    for (int rbw = 0; rbw < 256; ++rbw) {
        unsigned f[2] = { fw0[rbw], fw1[rbw] };
        // A fragment (contiguous K, matching B): lane m: K={0..15}; lane m+16:
        // K={16..31}. v16h element e -> K(e)=e+hi*16.
        // Packed pair ep holds elements {2ep, 2ep+1} -> bits {K0, K0+1}.
        AF a[2];
        #pragma unroll
        for (int t = 0; t < 2; ++t)
            #pragma unroll
            for (int ep = 0; ep < 8; ++ep) {
                int K0 = 2 * ep + hi * 16;
                unsigned lo  = ((f[t] >> K0)       & 1u) ? 0x3C00u     : 0u;
                unsigned hi2 = ((f[t] >> (K0 + 1)) & 1u) ? 0x3C000000u : 0u;
                a[t].u[ep] = lo | hi2;
            }
        // B fragments: lane n holds col n, 16 contiguous K halves (transposed).
        const _Float16* bbase = BmatT + (size_t)m * R_N + rbw * 32 + hi * 16;
        #pragma unroll
        for (int nt = 0; nt < 5; ++nt) {
            v16h bf = *(const v16h*)(bbase + (size_t)nt * 16 * R_N);
            #pragma unroll
            for (int t = 0; t < 2; ++t)
                acc[t][nt] = __builtin_amdgcn_wmma_f32_16x16x32_f16(
                                 false, a[t].h, false, bf, (short)0,
                                 acc[t][nt], false, false);
        }
    }

    // Scatter C/D layout (VGPR j -> row j + hi*8, N = nt*16 + lane%16) to LDS.
    #pragma unroll
    for (int t = 0; t < 2; ++t)
        #pragma unroll
        for (int nt = 0; nt < 5; ++nt)
            #pragma unroll
            for (int j = 0; j < 8; ++j)
                ldsS[wave][t * 16 + j + hi * 8][nt * 16 + m] = acc[t][nt][j];
    __syncthreads();

    // Epilogue: one row per lane. q=exp(S[:,0:64]), w=exp(S[:,64]).
    {
        float* Srow = ldsS[wave][lane];
        float  w    = expf(Srow[K_N]);
        float  tot  = w;
        for (int n = 0; n < K_N; ++n) {
            float belief = fmaxf(expf(Srow[n]) - w, 0.0f);
            Srow[n] = belief;
            tot += belief;
        }
        float invT = 1.0f / fmaxf(tot, EPSF);
        float* orow = out + (size_t)(sampleBase + lane) * K_N;
        for (int n = 0; n < K_N; ++n)
            orow[n] = (Srow[n] + w * prior[n]) * invT;
    }
}

// ---------------------------------------------------------------------------
extern "C" void kernel_launch(void* const* d_in, const int* in_sizes, int n_in,
                              void* d_out, int out_size, void* d_ws, size_t ws_size,
                              hipStream_t stream) {
    const float* X         = (const float*)d_in[0];
    const float* lit_value = (const float*)d_in[1];
    const float* lit2rule  = (const float*)d_in[2];
    // d_in[3] (rule_len) unused: fired == AND of the rule's distinct literals.
    const float* params    = (const float*)d_in[4];
    const float* prior     = (const float*)d_in[5];
    const int*   feat      = (const int*)d_in[6];
    const int*   opc       = (const int*)d_in[7];
    float*       out       = (float*)d_out;

    char* ws = (char*)d_ws;
    size_t off = 0;
    unsigned* truthBits  = (unsigned*)(ws + off); off += (size_t)B_N * (L_N / 32) * 4; // 4 MB
    unsigned* cnt        = (unsigned*)(ws + off); off += 64 << 10;                     // 32 KB (+pad)
    unsigned* tmp        = (unsigned*)(ws + off); off += 128 << 10;                    // 96 KB (+pad)
    uint2*    packed     = (uint2*)   (ws + off); off += 64 << 10;                     // 64 KB
    _Float16* BmatT      = (_Float16*)(ws + off); off += 2 << 20;                      // 1.25 MB (+pad)
    unsigned* firedWords = (unsigned*)(ws + off);                                      // 8 MB

    zero_counts   <<<R_N / 256, 256, 0, stream>>>(cnt);
    truth_kernel  <<<(B_N * (L_N / 32)) / 256, 256, 0, stream>>>(X, lit_value, feat, opc, truthBits);
    extract_kernel<<<(L_N * (size_t)R_N) / 256, 256, 0, stream>>>(lit2rule, cnt, tmp);
    pack_kernel   <<<R_N / 256, 256, 0, stream>>>(cnt, tmp, packed);
    softmax_kernel<<<R_N / 256, 256, 0, stream>>>(params, BmatT);
    fired_kernel  <<<B_N, 256, 0, stream>>>(truthBits, packed, firedWords);
    ds_gemm_kernel<<<B_N / 128, 128, 0, stream>>>(firedWords, BmatT, prior, out);
}
